// MLPGraphAttentionLayer_4784593567908
// MI455X (gfx1250) — matrix operand order repaired and checked by reading, not compiled, not run on hardware
//
#include <hip/hip_runtime.h>
#include <hip/hip_bf16.h>

#define ALPHA_S 0.2f
#define NEG_INF_V -9e15f
#define BB 2
#define NN 4096
#define FD 256
#define EMBD 16
#define NSPLIT_A 4
#define NSLICE_B 2

#if defined(__has_builtin)
# if __has_builtin(__builtin_amdgcn_tensor_load_to_lds) && __has_builtin(__builtin_amdgcn_s_wait_tensorcnt)
#  define HAVE_TDM 1
# endif
#endif
#ifndef HAVE_TDM
# define HAVE_TDM 0
#endif

typedef _Float16     v16h __attribute__((ext_vector_type(16)));
typedef _Float16     v8h  __attribute__((ext_vector_type(8)));
typedef float        v8f  __attribute__((ext_vector_type(8)));
typedef float        v4f  __attribute__((ext_vector_type(4)));
typedef unsigned int v4u  __attribute__((ext_vector_type(4)));
typedef int          v4i  __attribute__((ext_vector_type(4)));
typedef int          v8i  __attribute__((ext_vector_type(8)));

__device__ __forceinline__ float lrelu_f(float x){ return x > 0.f ? x : ALPHA_S * x; }

// ---- WMMA fragment loaders (gfx1250 wave32 layouts, 16x16x32 f16) ----
// A (16x32 f16): lane L holds row M=L&15; K = (idx>>3)*16 + ((L>=16)?8:0) + (idx&7)
__device__ __forceinline__ v16h load_a_frag_f32(const float* A, int lda, int m0, int k0, int lane){
  const float* p = A + (size_t)(m0 + (lane & 15)) * lda + k0 + ((lane >> 4) & 1) * 8;
  v4f x0 = *(const v4f*)(p);      v4f x1 = *(const v4f*)(p + 4);
  v4f x2 = *(const v4f*)(p + 16); v4f x3 = *(const v4f*)(p + 20);
  v16h a;
#pragma unroll
  for (int i = 0; i < 4; ++i){
    a[i]      = (_Float16)x0[i];
    a[4 + i]  = (_Float16)x1[i];
    a[8 + i]  = (_Float16)x2[i];
    a[12 + i] = (_Float16)x3[i];
  }
  return a;
}
__device__ __forceinline__ v16h load_a_frag_f16(const _Float16* A, int lda, int m0, int k0, int lane){
  const _Float16* p = A + (size_t)(m0 + (lane & 15)) * lda + k0 + ((lane >> 4) & 1) * 8;
  v8h lo = *(const v8h*)(p);
  v8h hi = *(const v8h*)(p + 16);
  v16h a;
#pragma unroll
  for (int i = 0; i < 8; ++i){ a[i] = lo[i]; a[8 + i] = hi[i]; }
  return a;
}
// B (32x16): lane L holds column N=L&15; K = (idx>>3)*16 + ((L>=16)?8:0) + (idx&7)
__device__ __forceinline__ v16h load_b_frag_f32(const float* Bw, int ldb, int k0, int n0, int lane){
  const float* p = Bw + (size_t)(k0 + ((lane >> 4) & 1) * 8) * ldb + n0 + (lane & 15);
  v16h b;
#pragma unroll
  for (int i = 0; i < 8; ++i){
    b[i]     = (_Float16)p[(size_t)i * ldb];
    b[8 + i] = (_Float16)p[(size_t)(16 + i) * ldb];
  }
  return b;
}
__device__ __forceinline__ v16h load_b_frag_f16(const _Float16* Bw, int ldb, int k0, int n0, int lane){
  const _Float16* p = Bw + (size_t)(k0 + ((lane >> 4) & 1) * 8) * ldb + n0 + (lane & 15);
  v16h b;
#pragma unroll
  for (int i = 0; i < 8; ++i){ b[i] = p[(size_t)i * ldb]; b[8 + i] = p[(size_t)(16 + i) * ldb]; }
  return b;
}
__device__ __forceinline__ v8f zero8(){
  v8f z;
#pragma unroll
  for (int i = 0; i < 8; ++i) z[i] = 0.f;
  return z;
}

#if HAVE_TDM
__device__ __forceinline__ unsigned int lds_offset_of(const void* p){
  return (unsigned int)(unsigned long long)(uintptr_t)p;  // low 32 bits of flat shared addr = LDS offset
}
// TDM: async DMA of a 2D tile (tile_dim0=32 contiguous f16 along j, tile_dim1=256 k-rows,
// row stride NN elements) from global hT into LDS. D# per CDNA5 ISA §8.3/8.4.
__device__ __forceinline__ void tdm_load_hT(const _Float16* gsrc, unsigned int ldsoff){
  unsigned long long ga = (unsigned long long)(uintptr_t)gsrc;
  v4u g0;
  g0[0] = 1u;                                   // count=1, user mode
  g0[1] = ldsoff;                               // lds_addr
  g0[2] = (unsigned int)(ga & 0xFFFFFFFFull);   // global_addr[31:0]
  g0[3] = (unsigned int)((ga >> 32) & 0x01FFFFFFull) | 0x80000000u; // ga[56:32] | type=2
  v8i g1;
  g1[0] = 1 << 16;                              // data_size=1 (2 bytes), mask=0
  g1[1] = (int)(((unsigned)NN & 0xFFFFu) << 16);// tensor_dim0[15:0]=4096
  g1[2] = (int)(((unsigned)FD & 0xFFFFu) << 16);// tensor_dim0[31:16]=0 | tensor_dim1[15:0]=256
  g1[3] = (int)(32u << 16);                     // tensor_dim1[31:16]=0 | tile_dim0=32
  g1[4] = 256;                                  // tile_dim1=256, tile_dim2=0
  g1[5] = NN;                                   // tensor_dim0_stride=4096
  g1[6] = 0;
  g1[7] = 0;
  v4i gz; gz[0]=0; gz[1]=0; gz[2]=0; gz[3]=0;
#if __clang_major__ >= 23
  v8i gz8; for (int i = 0; i < 8; ++i) gz8[i] = 0;
  __builtin_amdgcn_tensor_load_to_lds(g0, g1, gz, gz, gz8, 0);
#else
  __builtin_amdgcn_tensor_load_to_lds(g0, g1, gz, gz, 0);
#endif
}
#endif

// ---- Generic WMMA GEMM: C = act(A[M,K] @ Bw[K,Nc] + bias), 16x64 tile per wave ----
template<int DO_LRELU, int OUT32, int OUT16>
__global__ __launch_bounds__(256)
void gemm_wmma(const float* __restrict__ A, const float* __restrict__ Bw,
               const float* __restrict__ bias, float* __restrict__ Cout,
               _Float16* __restrict__ Chalf, int M, int Nc, int K){
  int wave = (blockIdx.x * blockDim.x + threadIdx.x) >> 5;
  int lane = threadIdx.x & 31;
  int ng = Nc >> 6;                      // groups of 64 columns
  int total = (M >> 4) * ng;
  if (wave >= total) return;
  int mt = wave / ng, nt = wave % ng;
  int m0 = mt << 4, n0 = nt << 6;
  v8f acc[4];
#pragma unroll
  for (int t = 0; t < 4; ++t) acc[t] = zero8();
  for (int k = 0; k < K; k += 32){
    v16h a = load_a_frag_f32(A, K, m0, k, lane);
#pragma unroll
    for (int t = 0; t < 4; ++t){
      v16h b = load_b_frag_f32(Bw, Nc, k, n0 + t * 16, lane);
      acc[t] = __builtin_amdgcn_wmma_f32_16x16x32_f16(false, a, false, b, (short)0, acc[t], false, false);
    }
  }
  int ncol = lane & 15;
  int mofs = (lane >> 4) << 3;
#pragma unroll
  for (int t = 0; t < 4; ++t){
    float bval = bias ? bias[n0 + t * 16 + ncol] : 0.f;
#pragma unroll
    for (int g = 0; g < 8; ++g){
      float v = acc[t][g] + bval;
      if (DO_LRELU) v = lrelu_f(v);
      size_t off = (size_t)(m0 + mofs + g) * Nc + n0 + t * 16 + ncol;
      if (OUT32) Cout[off]  = v;
      if (OUT16) Chalf[off] = (_Float16)v;
    }
  }
}

// ---- prep kernels ----
__global__ void conv_f16_kernel(const float* __restrict__ src, _Float16* __restrict__ dst, int n){
  int i = blockIdx.x * blockDim.x + threadIdx.x;
  if (i < n) dst[i] = (_Float16)src[i];
}
__global__ void transpose_conv_f16(const float* __restrict__ h, _Float16* __restrict__ hT){
  size_t idx = (size_t)blockIdx.x * blockDim.x + threadIdx.x;
  if (idx >= (size_t)BB * NN * FD) return;
  int k = (int)(idx & (FD - 1));
  size_t row = idx >> 8;           // b*N + n
  int n = (int)(row & (NN - 1));
  int b = (int)(row >> 12);
  hT[((size_t)b * FD + k) * NN + n] = (_Float16)h[idx];
}
__global__ void compute_axay(const float* __restrict__ h, const float* __restrict__ a1,
                             const float* __restrict__ a2, float* __restrict__ ax,
                             float* __restrict__ ay){
  int r = blockIdx.x * blockDim.x + threadIdx.x;
  if (r >= BB * NN) return;
  const float* hr = h + (size_t)r * FD;
  float s1 = 0.f, s2 = 0.f;
  for (int k = 0; k < FD; ++k){ float v = hr[k]; s1 += v * a1[k]; s2 += v * a2[k]; }
  ax[r] = s1; ay[r] = s2;
}

// ---- attention pass A: masked online max/sum of logits (flash-style, TDM-fed) ----
__global__ __launch_bounds__(32)
void attn_pass_a(const _Float16* __restrict__ tq, const _Float16* __restrict__ hT,
                 const float* __restrict__ ax, const float* __restrict__ ay,
                 const int* __restrict__ adj, float* __restrict__ m_part,
                 float* __restrict__ l_part){
#if HAVE_TDM
  __shared__ _Float16 hbuf[2][FD * 32];   // double-buffered 256x32 f16 tile (2 x 16KB)
#endif
  const int lane = threadIdx.x;
  int wid = blockIdx.x;
  const int it_per_b = NN / 16;
  int split = wid % NSPLIT_A; wid /= NSPLIT_A;
  int it = wid % it_per_b;
  int b  = wid / it_per_b;
  const int i0 = it * 16;
  const int hi8 = (lane >> 4) << 3;
  const int jlen = NN / NSPLIT_A;
  const int j0 = split * jlen;

  v16h av[8];
#pragma unroll
  for (int kc = 0; kc < 8; ++kc)
    av[kc] = load_a_frag_f16(tq + (size_t)b * NN * FD, FD, i0, kc * 32, lane);

  float ayv[8], mrun[8], lrun[8];
#pragma unroll
  for (int g = 0; g < 8; ++g){
    ayv[g] = ay[(size_t)b * NN + i0 + hi8 + g];
    mrun[g] = -3e38f; lrun[g] = 0.f;
  }
  const _Float16* hTb = hT + (size_t)b * FD * NN;
  const int* adj0 = adj + ((size_t)b * NN + i0 + hi8) * NN;

#if HAVE_TDM
  if (blockIdx.x == 0x7FFFFFFFu) hbuf[0][0] = (_Float16)0.f;  // keep LDS analysis honest
  unsigned int lds0 = lds_offset_of(&hbuf[0][0]);
  unsigned int lds1 = lds_offset_of(&hbuf[1][0]);
  tdm_load_hT(hTb + j0, lds0);
#endif
  int buf = 0;
  for (int j = j0; j < j0 + jlen; j += 32){
#if HAVE_TDM
    if (j + 32 < j0 + jlen){
      tdm_load_hT(hTb + j + 32, buf ? lds0 : lds1);
      __builtin_amdgcn_s_wait_tensorcnt(1);
    } else {
      __builtin_amdgcn_s_wait_tensorcnt(0);
    }
    asm volatile("" ::: "memory");
#endif
#pragma unroll
    for (int tj = 0; tj < 2; ++tj){
      int jb = j + tj * 16;
      v8f s = zero8();
#pragma unroll
      for (int kc = 0; kc < 8; ++kc){
#if HAVE_TDM
        const _Float16* p = &hbuf[buf][(kc * 32 + ((lane >> 4) << 3)) * 32 + tj * 16 + (lane & 15)];
        v16h bf;
#pragma unroll
        for (int i = 0; i < 8; ++i){ bf[i] = p[i * 32]; bf[8 + i] = p[(16 + i) * 32]; }
#else
        v16h bf = load_b_frag_f16(hTb, NN, kc * 32, jb, lane);
#endif
        s = __builtin_amdgcn_wmma_f32_16x16x32_f16(false, av[kc], false, bf, (short)0, s, false, false);
      }
      const int jc = jb + (lane & 15);
      const float axv = ax[(size_t)b * NN + jc];
#pragma unroll
      for (int g = 0; g < 8; ++g){
        float e = lrelu_f(s[g] + axv + ayv[g]);
        float val = (adj0[(size_t)g * NN + jc] > 0) ? e : NEG_INF_V;
        float mnew = fmaxf(mrun[g], val);
        lrun[g] = lrun[g] * __expf(mrun[g] - mnew) + __expf(val - mnew);
        mrun[g] = mnew;
      }
    }
    buf ^= 1;
  }
#pragma unroll
  for (int g = 0; g < 8; ++g){
    float m = mrun[g], l = lrun[g];
    for (int d = 1; d < 16; d <<= 1){
      float mo = __shfl_xor(m, d, 32);
      float lo = __shfl_xor(l, d, 32);
      float mn = fmaxf(m, mo);
      l = l * __expf(m - mn) + lo * __expf(mo - mn);
      m = mn;
    }
    if ((lane & 15) == 0){
      size_t r = (size_t)b * NN + i0 + hi8 + g;
      m_part[r * NSPLIT_A + split] = m;
      l_part[r * NSPLIT_A + split] = l;
    }
  }
}

__global__ void combine_ml(const float* __restrict__ m_part, const float* __restrict__ l_part,
                           float* __restrict__ Mrow, float* __restrict__ Lrow){
  int r = blockIdx.x * blockDim.x + threadIdx.x;
  if (r >= BB * NN) return;
  float m = -3e38f;
  for (int s = 0; s < NSPLIT_A; ++s) m = fmaxf(m, m_part[(size_t)r * NSPLIT_A + s]);
  float l = 0.f;
  for (int s = 0; s < NSPLIT_A; ++s)
    l += l_part[(size_t)r * NSPLIT_A + s] * __expf(m_part[(size_t)r * NSPLIT_A + s] - m);
  Mrow[r] = m; Lrow[r] = fmaxf(l, 1e-30f);
}

// ---- attention pass B: h_prime = softmax(S) @ feat, fused; hT streamed by TDM into LDS ----
__global__ __launch_bounds__(32)
void attn_pass_b(const _Float16* __restrict__ tq, const _Float16* __restrict__ hT,
                 const _Float16* __restrict__ featH, const float* __restrict__ ax,
                 const float* __restrict__ ay, const int* __restrict__ adj,
                 const float* __restrict__ Mrow, const float* __restrict__ Lrow,
                 float* __restrict__ h_prime){
#if HAVE_TDM
  __shared__ _Float16 hbuf[2][FD * 32];   // double-buffered 256x32 f16 tile (2 x 16KB)
#endif
  __shared__ float pl[32][17];
  const int lane = threadIdx.x;
  int wid = blockIdx.x;
  const int it_per_b = NN / 16;
  int slice = wid % NSLICE_B; wid /= NSLICE_B;
  int it = wid % it_per_b;
  int b  = wid / it_per_b;
  const int i0 = it * 16;
  const int hi8 = (lane >> 4) << 3;
  const int n0 = slice * (FD / NSLICE_B);
  const int NCH = FD / NSLICE_B / 16;     // 8 chunks of 16 columns

  v16h av[8];
#pragma unroll
  for (int kc = 0; kc < 8; ++kc)
    av[kc] = load_a_frag_f16(tq + (size_t)b * NN * FD, FD, i0, kc * 32, lane);

  float ayv[8], Mv[8], Li[8];
#pragma unroll
  for (int g = 0; g < 8; ++g){
    size_t r = (size_t)b * NN + i0 + hi8 + g;
    ayv[g] = ay[r]; Mv[g] = Mrow[r]; Li[g] = 1.f / Lrow[r];
  }
  v8f o[8];
#pragma unroll
  for (int c = 0; c < 8; ++c) o[c] = zero8();

  const _Float16* hTb = hT + (size_t)b * FD * NN;
  const _Float16* fb  = featH + (size_t)b * NN * FD;
  const int* adj0 = adj + ((size_t)b * NN + i0 + hi8) * NN;

#if HAVE_TDM
  if (blockIdx.x == 0x7FFFFFFFu) hbuf[0][0] = (_Float16)0.f;  // keep LDS analysis honest
  unsigned int lds0 = lds_offset_of(&hbuf[0][0]);
  unsigned int lds1 = lds_offset_of(&hbuf[1][0]);
  tdm_load_hT(hTb, lds0);
#endif
  int buf = 0;
  for (int j = 0; j < NN; j += 32){
#if HAVE_TDM
    if (j + 32 < NN){
      tdm_load_hT(hTb + j + 32, buf ? lds0 : lds1);
      __builtin_amdgcn_s_wait_tensorcnt(1);
    } else {
      __builtin_amdgcn_s_wait_tensorcnt(0);
    }
    asm volatile("" ::: "memory");
#endif
#pragma unroll
    for (int tj = 0; tj < 2; ++tj){
      int jb = j + tj * 16;
      v8f s = zero8();
#pragma unroll
      for (int kc = 0; kc < 8; ++kc){
#if HAVE_TDM
        // tile layout in LDS: [k=0..255][j_local=0..31]; lane holds column j = tj*16+(lane&15), k in table order
        const _Float16* p = &hbuf[buf][(kc * 32 + ((lane >> 4) << 3)) * 32 + tj * 16 + (lane & 15)];
        v16h bf;
#pragma unroll
        for (int i = 0; i < 8; ++i){ bf[i] = p[i * 32]; bf[8 + i] = p[(16 + i) * 32]; }
#else
        v16h bf = load_b_frag_f16(hTb, NN, kc * 32, jb, lane);
#endif
        s = __builtin_amdgcn_wmma_f32_16x16x32_f16(false, av[kc], false, bf, (short)0, s, false, false);
      }
      int jc = jb + (lane & 15);
      float axv = ax[(size_t)b * NN + jc];
#pragma unroll
      for (int g = 0; g < 8; ++g){
        float e = lrelu_f(s[g] + axv + ayv[g]);
        float val = (adj0[(size_t)g * NN + jc] > 0) ? e : NEG_INF_V;
        float p = __expf(val - Mv[g]) * Li[g];
        pl[tj * 16 + (lane & 15)][hi8 + g] = p;
      }
    }
    __syncthreads();
    // C-layout -> A-layout re-stage through LDS
    v16h pa;
#pragma unroll
    for (int idx = 0; idx < 16; ++idx){
      int jl = ((idx >> 3) << 4) + ((lane >= 16) ? 8 : 0) + (idx & 7);
      pa[idx] = (_Float16)pl[jl][lane & 15];
    }
    __syncthreads();
#pragma unroll
    for (int c = 0; c < NCH; ++c){
      v16h bf = load_b_frag_f16(fb, FD, j, n0 + c * 16, lane);
      o[c] = __builtin_amdgcn_wmma_f32_16x16x32_f16(false, pa, false, bf, (short)0, o[c], false, false);
    }
    buf ^= 1;
  }
#pragma unroll
  for (int c = 0; c < NCH; ++c)
#pragma unroll
    for (int g = 0; g < 8; ++g)
      h_prime[((size_t)b * NN + i0 + hi8 + g) * FD + n0 + c * 16 + (lane & 15)] = o[c][g];
}

// ---- embedding layers + concat (tiny, VALU) ----
__global__ void emb_concat(const float* __restrict__ feat, const float* __restrict__ h_prime,
                           const float* __restrict__ e1w, const float* __restrict__ e1b,
                           const float* __restrict__ e2w, const float* __restrict__ e2b,
                           float* __restrict__ cc){
  int t = blockIdx.x * blockDim.x + threadIdx.x;
  if (t >= BB * NN * 2 * EMBD) return;
  int r = t >> 5, u = t & 31;
  const float* src = (u < EMBD) ? feat : h_prime;
  const float* w   = (u < EMBD) ? e1w  : e2w;
  const float* bb  = (u < EMBD) ? e1b  : e2b;
  int uc = u & (EMBD - 1);
  float s = bb[uc];
  const float* row = src + (size_t)r * FD;
  for (int k = 0; k < FD; ++k) s += row[k] * w[k * EMBD + uc];
  cc[(size_t)r * 2 * EMBD + u] = lrelu_f(s);
}

extern "C" void kernel_launch(void* const* d_in, const int* in_sizes, int n_in,
                              void* d_out, int out_size, void* d_ws, size_t ws_size,
                              hipStream_t stream){
  const float* feat = (const float*)d_in[0];
  const int*   adj  = (const int*)d_in[1];
  const float* W    = (const float*)d_in[2];
  const float* a1   = (const float*)d_in[3];
  const float* a2   = (const float*)d_in[4];
  const float* a12  = (const float*)d_in[5];
  const float* e1w  = (const float*)d_in[6];
  const float* e1b  = (const float*)d_in[7];
  const float* e2w  = (const float*)d_in[8];
  const float* e2b  = (const float*)d_in[9];
  const float* i0w  = (const float*)d_in[10];
  const float* i0b  = (const float*)d_in[11];
  const float* i1w  = (const float*)d_in[12];
  const float* i1b  = (const float*)d_in[13];
  float* out = (float*)d_out;

  const size_t M = (size_t)BB * NN;         // 8192 rows
  char* ws = (char*)d_ws;
  size_t off = 0;
  auto alloc = [&](size_t bytes){ void* p = ws + off; off += (bytes + 255) & ~(size_t)255; return p; };
  float*    h_f32    = (float*)alloc(M * FD * 4);
  float*    h_prime  = (float*)alloc(M * FD * 4);
  float*    c1       = (float*)alloc(M * FD * 4);
  _Float16* t_f16    = (_Float16*)alloc(M * FD * 2);
  _Float16* hT_f16   = (_Float16*)alloc(M * FD * 2);
  _Float16* feat_f16 = (_Float16*)alloc(M * FD * 2);
  float*    cc       = (float*)alloc(M * 2 * EMBD * 4);
  float*    ax       = (float*)alloc(M * 4);
  float*    ay       = (float*)alloc(M * 4);
  float*    m_part   = (float*)alloc(M * NSPLIT_A * 4);
  float*    l_part   = (float*)alloc(M * NSPLIT_A * 4);
  float*    Mrow     = (float*)alloc(M * 4);
  float*    Lrow     = (float*)alloc(M * 4);

  // 1. h = feat @ W          (8192x256 x 256x256), 16x64 tiles -> 2048 waves
  gemm_wmma<0,1,0><<<256, 256, 0, stream>>>(feat, W, nullptr, h_f32, nullptr, (int)M, FD, FD);
  // 2. t = h @ a12 (f16 output, A-operand of the logits GEMM)
  gemm_wmma<0,0,1><<<256, 256, 0, stream>>>(h_f32, a12, nullptr, nullptr, t_f16, (int)M, FD, FD);
  // 3. hT (f16), feat (f16), ax/ay
  transpose_conv_f16<<<(int)(M * FD / 256), 256, 0, stream>>>(h_f32, hT_f16);
  conv_f16_kernel<<<(int)(M * FD / 256), 256, 0, stream>>>(feat, feat_f16, (int)(M * FD));
  compute_axay<<<(int)(M / 256), 256, 0, stream>>>(h_f32, a1, a2, ax, ay);
  // 4. flash pass A: per-row masked max & sum (TDM double-buffered hT tiles)
  attn_pass_a<<<BB * (NN / 16) * NSPLIT_A, 32, 0, stream>>>(t_f16, hT_f16, ax, ay, adj, m_part, l_part);
  combine_ml<<<(int)(M / 256), 256, 0, stream>>>(m_part, l_part, Mrow, Lrow);
  // 5. flash pass B: h_prime = softmax @ feat (TDM double-buffered hT tiles)
  attn_pass_b<<<BB * (NN / 16) * NSLICE_B, 32, 0, stream>>>(t_f16, hT_f16, feat_f16, ax, ay, adj,
                                                            Mrow, Lrow, h_prime);
  // 6. BiInteraction MLP
  emb_concat<<<(int)(M * 2 * EMBD / 256), 256, 0, stream>>>(feat, h_prime, e1w, e1b, e2w, e2b, cc);
  gemm_wmma<1,1,0><<<256, 256, 0, stream>>>(cc, i0w, i0b, c1, nullptr, (int)M, FD, 2 * EMBD);
  gemm_wmma<1,1,0><<<256, 256, 0, stream>>>(c1, i1w, i1b, out, nullptr, (int)M, FD, FD);
}
